// BaseModel_28716151341603
// MI455X (gfx1250) — matrix operand order repaired and checked by reading, not compiled, not run on hardware
//
#include <hip/hip_runtime.h>
#include <math.h>

// ---------------- CDNA5 WMMA types ----------------
typedef __attribute__((ext_vector_type(16))) __bf16 v16bf;
typedef __attribute__((ext_vector_type(8)))  float  v8f;

#define NWG 28
#define TSTEPS 95

__device__ __forceinline__ unsigned f2bf(float f) {
  unsigned u = __float_as_uint(f);
  unsigned r = ((u >> 16) & 1u) + 0x7fffu;   // round-to-nearest-even
  return (u + r) >> 16;
}
__device__ __forceinline__ unsigned pack2(float lo, float hi) {
  return f2bf(lo) | (f2bf(hi) << 16);
}
__device__ __forceinline__ float sigm(float x) { return 1.f / (1.f + expf(-x)); }
__device__ __forceinline__ int imax(int a, int b) { return a > b ? a : b; }
__device__ __forceinline__ int imin(int a, int b) { return a < b ? a : b; }

union FragU { unsigned u[8]; uint4 q[2]; v16bf v; };

// ---- fragment-major layouts: every lane's 16 bf16 are 8 consecutive u32 ----
// A storage: [row m][ks][half][8]  (half = lane>>4)
// B storage: [panel][ks][lane][8]  (lane = half*16 + n)
// map logical k-pair index kp (pair-packed row) -> fragment-major slot
__device__ __forceinline__ int amap(int m, int KS, int kp) {
  int ks = kp >> 4, p = kp & 15;
  int hig = (p >> 2) & 1;                 // ISA 7.12.2 A-matrix pair pattern
  int j = (p & 3) | ((p >> 3) << 2);
  return ((m * KS + ks) * 2 + hig) * 8 + j;
}
__device__ __forceinline__ v16bf load_af(const unsigned* A, int KS, int mbase, int ks) {
  int lane = threadIdx.x & 31;
  const uint4* q = (const uint4*)(A + (((mbase + (lane & 15)) * KS + ks) * 2 + (lane >> 4)) * 8);
  FragU f; f.q[0] = q[0]; f.q[1] = q[1];
  return f.v;
}
__device__ __forceinline__ v16bf load_bf(const unsigned* Wf, int ks) {
  const uint4* q = (const uint4*)(Wf + (ks * 32 + (threadIdx.x & 31)) * 8);
  FragU f; f.q[0] = q[0]; f.q[1] = q[1];
  return f.v;
}
__device__ __forceinline__ v8f wmma_bf16(v16bf a, v16bf b, v8f c) {
  return __builtin_amdgcn_wmma_f32_16x16x32_bf16(false, a, false, b, (short)0, c, false, false);
}
// D layout: lanes<16 -> rows v, lanes>=16 -> rows v+8; col = lane&15
__device__ __forceinline__ void store_tile32(float* G, int ldg, int n0, const float* bias,
                                             const v8f& a0, const v8f& a1) {
  int lane = threadIdx.x & 31;
  int n = lane & 15, hi = (lane >> 4) & 1;
  float bv = bias[n0 + n];
#pragma unroll
  for (int v = 0; v < 8; ++v) {
    int m = v + 8 * hi;
    G[m * ldg + n0 + n]        = a0[v] + bv;
    G[(m + 16) * ldg + n0 + n] = a1[v] + bv;
  }
}

// sense-reversing grid barrier (bar[0]=count, bar[1]=generation)
__device__ __forceinline__ void gbar(unsigned* bar) {
  __syncthreads();
  if (threadIdx.x == 0) {
    __threadfence();
    volatile unsigned* gen = bar + 1;
    unsigned g = *gen;
    if (atomicAdd(bar, 1u) == NWG - 1) {
      bar[0] = 0;
      __threadfence();
      atomicAdd((unsigned*)(bar + 1), 1u);
    } else {
      while (*gen == g) __builtin_amdgcn_s_sleep(1);
    }
    __threadfence();
  }
  __syncthreads();
}

__device__ __forceinline__ float trk_th(const float* Gr, const float* tcr, int j) {
  float tcn = sigm(Gr[128 + j]) * tcr[j] + sigm(Gr[j]) * tanhf(Gr[256 + j]);
  return sigm(Gr[384 + j]) * tanhf(tcn);
}
// TreeLSTM elementwise; Grow = gates + b*1280 + j ; splits at 256 boundaries
__device__ __forceinline__ void comp_elem(const float* Grow, float cl, float cr,
                                          float bh, int red, float* nh, float* nc) {
  float c  = sigm(Grow[256]) * cl + sigm(Grow[512]) * cr + sigm(Grow[0]) * tanhf(Grow[1024]);
  float rh = sigm(Grow[768]) * tanhf(c);
  *nh = red ? rh : bh;
  *nc = red ? c  : 0.f;
}

struct WS {
  unsigned *trkW0p, *trkW1p, *compW0p, *compW1p;
  unsigned *encW0p, *encW1p, *mlpW1p;
  unsigned *Xe, *bufs0p, *hpack;
  float *bufs0, *bufs1;
  float *sh0, *sc0, *sh1, *sc1;
  float *thbuf, *tcbuf;       // [2 parity][2 layer][32][128]
  float *pend_h0, *pend_c0;   // [32][256]
  float *gtrk0, *gtrk1, *gcomp0, *gcomp1;
  float *hid;
  int *ptr0, *ptr1, *bp0, *bp1;
  unsigned *bar;
};

// ---------------- prep kernels ----------------
__global__ void k_zero(unsigned* p, unsigned n) {
  unsigned i = blockIdx.x * 256 + threadIdx.x;
  if (i < n) p[i] = 0u;
}
// pack f32 weights [K][N] -> bf16 fragment-major [panel][ks][lane][8]
__global__ void k_pack_w(const float* __restrict__ W, unsigned* __restrict__ Wp,
                         int K, int N, int KS) {
  int idx = blockIdx.x * 256 + threadIdx.x;
  if (idx >= (KS * 16) * N) return;
  int j = idx & 7, t = idx >> 3;
  int lane = t & 31; t >>= 5;
  int ks = t % KS, panel = t / KS;
  int k0 = ks * 32 + (j >> 2) * 16 + (lane >> 4) * 8 + (j & 3) * 2;       // B-matrix K pair pattern
  int col = panel * 16 + (lane & 15);
  float x0 = (k0 < K)     ? W[k0 * N + col]       : 0.f;
  float x1 = (k0 + 1 < K) ? W[(k0 + 1) * N + col] : 0.f;
  Wp[idx] = pack2(x0, x1);
}
__global__ void k_embed_pack(const int* __restrict__ tokens, const float* __restrict__ embed,
                             unsigned* __restrict__ Xe) {
  int idx = blockIdx.x * 256 + threadIdx.x;
  if (idx >= 1536 * 160) return;
  int m = idx / 160, kp = idx - m * 160, c = kp * 2;
  const float* row = embed + (size_t)tokens[m] * 300;
  float x0 = (c < 300)     ? row[c]     : 0.f;
  float x1 = (c + 1 < 300) ? row[c + 1] : 0.f;
  Xe[amap(m, 10, kp)] = pack2(x0, x1);
}
__global__ void k_pack_act(const float* __restrict__ A, unsigned* __restrict__ Ap) {
  int idx = blockIdx.x * 256 + threadIdx.x;
  if (idx >= 1536 * 128) return;
  int m = idx >> 7, kp = idx & 127;
  Ap[amap(m, 8, kp)] = pack2(A[m * 256 + kp * 2], A[m * 256 + kp * 2 + 1]);
}

// encoder GEMM: each wave computes a 32x16 tile, bf16 WMMA, f32 out + bias
__global__ __launch_bounds__(256)
void k_gemm(const unsigned* __restrict__ Ap, const unsigned* __restrict__ Wp,
            const float* __restrict__ bias, float* __restrict__ Cout,
            int M32, int NT, int KS, int N) {
  int wave = blockIdx.x * 8 + (threadIdx.x >> 5);
  if (wave >= M32 * NT) return;
  int mt = wave % M32, nt = wave / M32, n0 = nt * 16;
  const unsigned* Wf = Wp + (size_t)nt * KS * 256;
  v8f a0 = {0,0,0,0,0,0,0,0}, a1 = {0,0,0,0,0,0,0,0};
#pragma unroll 2
  for (int ks = 0; ks < KS; ++ks) {
    v16bf bf = load_bf(Wf, ks);
    a0 = wmma_bf16(load_af(Ap, KS, mt * 32,      ks), bf, a0);
    a1 = wmma_bf16(load_af(Ap, KS, mt * 32 + 16, ks), bf, a1);
  }
  int lane = threadIdx.x & 31, n = lane & 15, hi = (lane >> 4) & 1;
  float bv = bias[n0 + n];
#pragma unroll
  for (int v = 0; v < 8; ++v) {
    int m = mt * 32 + v + 8 * hi;
    Cout[m * N + n0 + n]        = a0[v] + bv;
    Cout[(m + 16) * N + n0 + n] = a1[v] + bv;
  }
}

// ---------------- persistent scan kernel ----------------
// 28 WGs x 256 thr. WGs 0-3: trk0, 4-7: trk1, 8-17: comp0, 18-27: comp1.
// Each wave owns one 16-column weight panel, LDS-resident for all 95 steps.
__global__ __launch_bounds__(256, 1)
void k_scan(WS w, const int* __restrict__ trans,
            const float* __restrict__ trk_b0, const float* __restrict__ trk_b1,
            const float* __restrict__ comp_b0, const float* __restrict__ comp_b1,
            const float* __restrict__ mlp_b1, const float* __restrict__ mlp_W2,
            const float* __restrict__ mlp_b2, float* __restrict__ out) {
  extern __shared__ unsigned smem[];
  const int wg = blockIdx.x, tid = threadIdx.x, wv = tid >> 5;

  int cls, ntile, KP, N;
  const unsigned* Wp; const float* bias; float* gates;
  if (wg < 4)       { cls = 0; KP = 448; N = 512;  Wp = w.trkW0p;  bias = trk_b0;  gates = w.gtrk0;  ntile = wg * 8 + wv; }
  else if (wg < 8)  { cls = 1; KP = 448; N = 512;  Wp = w.trkW1p;  bias = trk_b1;  gates = w.gtrk1;  ntile = (wg - 4) * 8 + wv; }
  else if (wg < 18) { cls = 2; KP = 320; N = 1280; Wp = w.compW0p; bias = comp_b0; gates = w.gcomp0; ntile = (wg - 8) * 8 + wv; }
  else              { cls = 3; KP = 448; N = 1280; Wp = w.compW1p; bias = comp_b1; gates = w.gcomp1; ntile = (wg - 18) * 8 + wv; }
  const int n0 = ntile * 16;
  const int KS = KP >> 4;
  const int layer = (cls == 1 || cls == 3) ? 1 : 0;

  int* mIdx = (int*)smem;                       // [0..95]: top/sec/buf indices
  unsigned* Xl = smem + 128;                    // activation staging, fragment-major
  unsigned* Wl = Xl + 32 * KP + wv * KP * 16;   // this wave's weight panel (fragment-major)

  // one-time: stage weight panel into LDS — contiguous uint4 copy
  {
    uint4* dq = (uint4*)Wl;
    const uint4* sq = (const uint4*)(Wp + (size_t)ntile * KS * 256);
    for (int i = tid & 31; i < KS * 64; i += 32) dq[i] = sq[i];
  }
  __syncthreads();

  const float* shp = layer ? w.sh1 : w.sh0;
  const float* bfp = layer ? w.bufs1 : w.bufs0;
  const int* ptrp = layer ? w.ptr1 : w.ptr0;
  const int* bpp  = layer ? w.bp1  : w.bp0;

  for (int t = 0; t < TSTEPS; ++t) {
    const int par = t & 1;
    if (tid < 32) {
      int p = ptrp[tid];
      mIdx[tid]      = imax(p - 1, 0);
      mIdx[32 + tid] = imax(p - 2, 0);
      mIdx[64 + tid] = imin(bpp[tid], 47);
    }
    __syncthreads();
    // ---- Phase 1: build X in LDS, trk full GEMM + comp low-K GEMM ----
    if (cls <= 1) { // X_trk = [b_h | top_h | sec_h | th_prev]
      const float* thp = w.thbuf + (par * 2 + layer) * 4096;
      for (int i = tid; i < 32 * 448; i += 256) {
        int b = i / 448, kp = i - b * 448, c = kp * 2;
        const float* r;
        if (c < 256)      r = bfp + (b * 48 + mIdx[64 + b]) * 256 + c;
        else if (c < 512) r = shp + (b * 49 + mIdx[b])      * 256 + (c - 256);
        else if (c < 768) r = shp + (b * 49 + mIdx[32 + b]) * 256 + (c - 512);
        else              r = thp + b * 128 + (c - 768);
        Xl[amap(b, 28, kp)] = pack2(r[0], r[1]);
      }
    } else {        // X_comp cols [sec_h | top_h] (th / ext filled later)
      for (int i = tid; i < 32 * 256; i += 256) {
        int b = i / 256, kp = i - b * 256, c = kp * 2;
        const float* r = (c < 256) ? (shp + (b * 49 + mIdx[32 + b]) * 256 + c)
                                   : (shp + (b * 49 + mIdx[b]) * 256 + (c - 256));
        Xl[amap(b, KS, kp)] = pack2(r[0], r[1]);
      }
    }
    __syncthreads();
    v8f acc0 = {0,0,0,0,0,0,0,0}, acc1 = {0,0,0,0,0,0,0,0};
    int ks1 = (cls <= 1) ? 28 : 16;
#pragma unroll 4
    for (int ks = 0; ks < ks1; ++ks) {
      v16bf bf = load_bf(Wl, ks);
      acc0 = wmma_bf16(load_af(Xl, KS, 0,  ks), bf, acc0);
      acc1 = wmma_bf16(load_af(Xl, KS, 16, ks), bf, acc1);
    }
    if (cls <= 1) store_tile32(gates, N, n0, bias, acc0, acc1);
    gbar(w.bar);

    // ---- Phase 2: th (redundant per comp WG) -> LDS cols, comp th-K GEMM ----
    if (cls >= 2) {
      const float* G = layer ? w.gtrk1 : w.gtrk0;
      const float* tco = w.tcbuf + (par * 2 + layer) * 4096;
      for (int i = tid; i < 32 * 64; i += 256) {
        int b = i / 64, pp = i - b * 64, j = pp * 2;
        const float* Gr = G + b * 512;
        const float* tr = tco + b * 128;
        Xl[amap(b, KS, 256 + pp)] = pack2(trk_th(Gr, tr, j), trk_th(Gr, tr, j + 1));
      }
      __syncthreads();
#pragma unroll
      for (int ks = 16; ks < 20; ++ks) {
        v16bf bf = load_bf(Wl, ks);
        acc0 = wmma_bf16(load_af(Xl, KS, 0,  ks), bf, acc0);
        acc1 = wmma_bf16(load_af(Xl, KS, 16, ks), bf, acc1);
      }
      if (cls == 2) store_tile32(gates, N, n0, bias, acc0, acc1);
    }
    if (wg == 0) { // commit th/tc (both layers) to opposite parity buffers
      for (int i = tid; i < 8192; i += 256) {
        int ly = i >> 12, r = i & 4095, b = r >> 7, j = r & 127;
        const float* Gr = (ly ? w.gtrk1 : w.gtrk0) + b * 512;
        float tco = w.tcbuf[(par * 2 + ly) * 4096 + b * 128 + j];
        float tcn = sigm(Gr[128 + j]) * tco + sigm(Gr[j]) * tanhf(Gr[256 + j]);
        w.tcbuf[((1 - par) * 2 + ly) * 4096 + b * 128 + j] = tcn;
        w.thbuf[((1 - par) * 2 + ly) * 4096 + b * 128 + j] = sigm(Gr[384 + j]) * tanhf(tcn);
      }
    }
    gbar(w.bar);

    // ---- Phase 3: ext (redundant comp0 elementwise) -> comp1 ext-K GEMM ----
    if (cls == 3) {
      for (int i = tid; i < 32 * 128; i += 256) {
        int b = i / 128, pp = i - b * 128, j = pp * 2;
        int p0 = w.ptr0[b], sec = imax(p0 - 2, 0), top = imax(p0 - 1, 0);
        int red = trans[b * 95 + t] == 1;
        const float* bh = w.bufs0 + (b * 48 + imin(w.bp0[b], 47)) * 256;
        float nh0, nc0, nh1, nc1;
        comp_elem(w.gcomp0 + b * 1280 + j,     w.sc0[(b * 49 + sec) * 256 + j],
                  w.sc0[(b * 49 + top) * 256 + j],     bh[j],     red, &nh0, &nc0);
        comp_elem(w.gcomp0 + b * 1280 + j + 1, w.sc0[(b * 49 + sec) * 256 + j + 1],
                  w.sc0[(b * 49 + top) * 256 + j + 1], bh[j + 1], red, &nh1, &nc1);
        Xl[amap(b, 28, 320 + pp)] = pack2(nh0, nh1);
      }
      __syncthreads();
#pragma unroll
      for (int ks = 20; ks < 28; ++ks) {
        v16bf bf = load_bf(Wl, ks);
        acc0 = wmma_bf16(load_af(Xl, KS, 0,  ks), bf, acc0);
        acc1 = wmma_bf16(load_af(Xl, KS, 16, ks), bf, acc1);
      }
      store_tile32(gates, N, n0, bias, acc0, acc1);
    }
    if (wg == 0) { // comp0 elementwise into pending buffers (commit next phase)
      for (int i = tid; i < 8192; i += 256) {
        int b = i >> 8, j = i & 255;
        int p0 = w.ptr0[b], sec = imax(p0 - 2, 0), top = imax(p0 - 1, 0);
        int red = trans[b * 95 + t] == 1;
        float bh = w.bufs0[(b * 48 + imin(w.bp0[b], 47)) * 256 + j];
        float nh, nc;
        comp_elem(w.gcomp0 + b * 1280 + j, w.sc0[(b * 49 + sec) * 256 + j],
                  w.sc0[(b * 49 + top) * 256 + j], bh, red, &nh, &nc);
        w.pend_h0[i] = nh; w.pend_c0[i] = nc;
      }
    }
    gbar(w.bar);

    // ---- Phase 4: commit layer0 stack (wg0); comp1 elementwise+commit (wg1) ----
    if (wg == 0) {
      for (int i = tid; i < 8192; i += 256) {
        int b = i >> 8, j = i & 255;
        int p0 = w.ptr0[b];
        int red = trans[b * 95 + t] == 1;
        int pos = imax(red ? p0 - 2 : p0, 0);
        w.sh0[(b * 49 + pos) * 256 + j] = w.pend_h0[i];
        w.sc0[(b * 49 + pos) * 256 + j] = w.pend_c0[i];
      }
      __syncthreads();
      if (tid < 32) {
        int tr = trans[tid * 95 + t];
        w.ptr0[tid] += (tr == 1) ? -1 : 1;
        w.bp0[tid]  += 1 - tr;
      }
    }
    if (wg == 1) {
      if (tid < 32) {
        int p1 = w.ptr1[tid];
        mIdx[tid] = imax(p1 - 1, 0); mIdx[32 + tid] = imax(p1 - 2, 0);
        mIdx[64 + tid] = imin(w.bp1[tid], 47);
      }
      __syncthreads();
      for (int i = tid; i < 8192; i += 256) {
        int b = i >> 8, j = i & 255;
        int red = trans[b * 95 + t] == 1;
        float bh = w.bufs1[(b * 48 + mIdx[64 + b]) * 256 + j];
        float nh, nc;
        comp_elem(w.gcomp1 + b * 1280 + j, w.sc1[(b * 49 + mIdx[32 + b]) * 256 + j],
                  w.sc1[(b * 49 + mIdx[b]) * 256 + j], bh, red, &nh, &nc);
        int p1 = w.ptr1[b];
        int pos = imax(red ? p1 - 2 : p1, 0);
        w.sh1[(b * 49 + pos) * 256 + j] = nh;
        w.sc1[(b * 49 + pos) * 256 + j] = nc;
      }
      __syncthreads();
      if (tid < 32) {
        int tr = trans[tid * 95 + t];
        w.ptr1[tid] += (tr == 1) ? -1 : 1;
        w.bp1[tid]  += 1 - tr;
      }
    }
    gbar(w.bar);
  }

  // ---- MLP head ----
  if (wg == 0) {
    if (tid < 32) mIdx[tid] = imax(w.ptr1[tid] - 1, 0);
    __syncthreads();
    for (int i = tid; i < 32 * 128; i += 256) {
      int b = i >> 7, kp = i & 127;
      const float* r = w.sh1 + (b * 49 + mIdx[b]) * 256 + kp * 2;
      w.hpack[amap(b, 8, kp)] = pack2(r[0], r[1]);
    }
  }
  gbar(w.bar);
  if (cls == 3 && ntile < 64) {
    const unsigned* Wf = w.mlpW1p + (size_t)ntile * 8 * 256;
    v8f a0 = {0,0,0,0,0,0,0,0}, a1 = {0,0,0,0,0,0,0,0};
#pragma unroll
    for (int ks = 0; ks < 8; ++ks) {
      v16bf bf = load_bf(Wf, ks);
      a0 = wmma_bf16(load_af(w.hpack, 8, 0,  ks), bf, a0);
      a1 = wmma_bf16(load_af(w.hpack, 8, 16, ks), bf, a1);
    }
    int lane = tid & 31, n = lane & 15, hi = (lane >> 4) & 1;
    float bv = mlp_b1[n0 + n];
#pragma unroll
    for (int v = 0; v < 8; ++v) {
      int m = v + 8 * hi;
      w.hid[m * 1024 + n0 + n]        = fmaxf(a0[v] + bv, 0.f);
      w.hid[(m + 16) * 1024 + n0 + n] = fmaxf(a1[v] + bv, 0.f);
    }
  }
  gbar(w.bar);
  if (wg == 0 && tid < 96) {
    int b = tid / 3, c = tid - b * 3;
    float s = mlp_b2[c];
    for (int k = 0; k < 1024; ++k) s += w.hid[b * 1024 + k] * mlp_W2[k * 3 + c];
    out[b * 3 + c] = s;
  }
}

// ---------------- host ----------------
extern "C" void kernel_launch(void* const* d_in, const int* in_sizes, int n_in,
                              void* d_out, int out_size, void* d_ws, size_t ws_size,
                              hipStream_t stream) {
  (void)in_sizes; (void)n_in; (void)out_size; (void)ws_size;
  const int*   tokens  = (const int*)d_in[0];
  const int*   trans   = (const int*)d_in[1];
  const float* embed   = (const float*)d_in[2];
  const float* enc_W0  = (const float*)d_in[3];
  const float* enc_b0  = (const float*)d_in[4];
  const float* enc_W1  = (const float*)d_in[5];
  const float* enc_b1  = (const float*)d_in[6];
  const float* trk_W0  = (const float*)d_in[7];
  const float* trk_b0  = (const float*)d_in[8];
  const float* trk_W1  = (const float*)d_in[9];
  const float* trk_b1  = (const float*)d_in[10];
  const float* comp_W0 = (const float*)d_in[11];
  const float* comp_b0 = (const float*)d_in[12];
  const float* comp_W1 = (const float*)d_in[13];
  const float* comp_b1 = (const float*)d_in[14];
  const float* mlp_W1  = (const float*)d_in[15];
  const float* mlp_b1  = (const float*)d_in[16];
  const float* mlp_W2  = (const float*)d_in[17];
  const float* mlp_b2  = (const float*)d_in[18];

  char* p = (char*)d_ws;
  auto take = [&](size_t bytes) -> char* {
    char* r = p; p += (bytes + 255) & ~(size_t)255; return r;
  };
  WS w;
  w.trkW0p  = (unsigned*)take(448 * 512 * 4);
  w.trkW1p  = (unsigned*)take(448 * 512 * 4);
  w.compW0p = (unsigned*)take(320 * 1280 * 4);
  w.compW1p = (unsigned*)take(448 * 1280 * 4);
  w.encW0p  = (unsigned*)take(160 * 256 * 4);
  w.encW1p  = (unsigned*)take(128 * 256 * 4);
  w.mlpW1p  = (unsigned*)take(128 * 1024 * 4);
  w.Xe      = (unsigned*)take(1536 * 160 * 4);
  w.bufs0   = (float*)take(1536 * 256 * 4);
  w.bufs0p  = (unsigned*)take(1536 * 128 * 4);
  w.bufs1   = (float*)take(1536 * 256 * 4);
  w.hpack   = (unsigned*)take(32 * 128 * 4);
  char* zstart = p;
  w.sh0     = (float*)take(32 * 49 * 256 * 4);
  w.sc0     = (float*)take(32 * 49 * 256 * 4);
  w.sh1     = (float*)take(32 * 49 * 256 * 4);
  w.sc1     = (float*)take(32 * 49 * 256 * 4);
  w.thbuf   = (float*)take(2 * 2 * 32 * 128 * 4);
  w.tcbuf   = (float*)take(2 * 2 * 32 * 128 * 4);
  w.pend_h0 = (float*)take(32 * 256 * 4);
  w.pend_c0 = (float*)take(32 * 256 * 4);
  w.gtrk0   = (float*)take(32 * 512 * 4);
  w.gtrk1   = (float*)take(32 * 512 * 4);
  w.gcomp0  = (float*)take(32 * 1280 * 4);
  w.gcomp1  = (float*)take(32 * 1280 * 4);
  w.hid     = (float*)take(32 * 1024 * 4);
  w.ptr0    = (int*)take(32 * 4);
  w.ptr1    = (int*)take(32 * 4);
  w.bp0     = (int*)take(32 * 4);
  w.bp1     = (int*)take(32 * 4);
  w.bar     = (unsigned*)take(256);
  unsigned zwords = (unsigned)((size_t)(p - zstart) / 4);

  auto g = [](int n) { return (n + 255) / 256; };
  // KS = Kpad/32 k-steps -> pass KS = KP/16 (pairs per 16) /... (KS here = K-steps of 32 = KP/16)
  k_pack_w<<<g(448 * 512), 256, 0, stream>>>(trk_W0, w.trkW0p, 896, 512, 28);
  k_pack_w<<<g(448 * 512), 256, 0, stream>>>(trk_W1, w.trkW1p, 896, 512, 28);
  k_pack_w<<<g(320 * 1280), 256, 0, stream>>>(comp_W0, w.compW0p, 640, 1280, 20);
  k_pack_w<<<g(448 * 1280), 256, 0, stream>>>(comp_W1, w.compW1p, 896, 1280, 28);
  k_pack_w<<<g(160 * 256), 256, 0, stream>>>(enc_W0, w.encW0p, 300, 256, 10);
  k_pack_w<<<g(128 * 256), 256, 0, stream>>>(enc_W1, w.encW1p, 256, 256, 8);
  k_pack_w<<<g(128 * 1024), 256, 0, stream>>>(mlp_W1, w.mlpW1p, 256, 1024, 8);
  k_zero<<<g((int)zwords), 256, 0, stream>>>((unsigned*)zstart, zwords);
  k_embed_pack<<<g(1536 * 160), 256, 0, stream>>>(tokens, embed, w.Xe);
  k_gemm<<<96, 256, 0, stream>>>(w.Xe, w.encW0p, enc_b0, w.bufs0, 48, 16, 10, 256);
  k_pack_act<<<g(1536 * 128), 256, 0, stream>>>(w.bufs0, w.bufs0p);
  k_gemm<<<96, 256, 0, stream>>>(w.bufs0p, w.encW1p, enc_b1, w.bufs1, 48, 16, 8, 256);
  // dynamic LDS: 128 misc + 32*448 X + 8*448*16 weight panels = 71808 u32 = 287232 B
  k_scan<<<NWG, 256, 287232, stream>>>(w, trans, trk_b0, trk_b1, comp_b0, comp_b1,
                                       mlp_b1, mlp_W2, mlp_b2, (float*)d_out);
}
